// _seblock_10892037062956
// MI455X (gfx1250) — matrix operand order repaired and checked by reading, not compiled, not run on hardware
//
#include <hip/hip_runtime.h>
#include <hip/hip_bf16.h>

// Problem constants (fixed by setup_inputs): B=4, C=64, H=64, W2=128,
// scale=2 -> w=64, bs=32, nw=2, Bn=16 blocks, N=2048 tokens/block.
// Workspace (12 MB): q_t [16][2048][64] bf16 (token-major),
//                    k_t [16][2048][64] bf16 (token-major),
//                    v   [16][64][2048] bf16 (channel-major).

typedef __bf16 bf16;
typedef __attribute__((ext_vector_type(4)))  bf16  v4bf;
typedef __attribute__((ext_vector_type(8)))  bf16  v8bf;
typedef __attribute__((ext_vector_type(16))) bf16  v16bf;
typedef __attribute__((ext_vector_type(4)))  float v4f;
typedef __attribute__((ext_vector_type(8)))  float v8f;

union V16 { v16bf v; v8bf h[2]; };

// A operand (16x32 bf16, M x K): storage row-major [row][k], stride in halves.
// lane<16: K = {k0..k0+7} u {k0+16..k0+23}; lane>=16: +8 on both chunks.
__device__ __forceinline__ v16bf load_matA(const bf16* base, int row, int k0,
                                           int stride, int lane) {
  const bf16* p = base + row * stride + k0 + ((lane < 16) ? 0 : 8);
  V16 u;
  u.h[0] = *(const v8bf*)(p);
  u.h[1] = *(const v8bf*)(p + 16);
  return u.v;
}
// B operand (32x16 bf16, K x N): storage holds B^T row-major [col][k].
// lane<16: K = {k0..k0+7} u {k0+16..k0+23}; lane>=16: +8 on both chunks.
__device__ __forceinline__ v16bf load_matB(const bf16* base, int col, int k0,
                                           int stride, int lane) {
  const bf16* p = base + col * stride + k0 + ((lane < 16) ? 0 : 8);
  V16 u;
  u.h[0] = *(const v8bf*)(p);
  u.h[1] = *(const v8bf*)(p + 16);
  return u.v;
}

__device__ __forceinline__ v8f wmma_bf16(v16bf a, v16bf b, v8f c) {
  return __builtin_amdgcn_wmma_f32_16x16x32_bf16(
      /*neg_a=*/false, a, /*neg_b=*/false, b,
      /*c_mod=*/(short)0, c, /*reuse_a=*/false, /*reuse_b=*/false);
}

// butterfly shuffle within the wave32 (stays inside 16-lane halves for mask<16)
__device__ __forceinline__ float xshf(float v, int mask) {
  int lane = (int)(threadIdx.x & 31);
  int i = __builtin_bit_cast(int, v);
  i = __builtin_amdgcn_ds_bpermute((((lane ^ mask) & 31) << 2), i);
  return __builtin_bit_cast(float, i);
}

// ---------------------------------------------------------------------------
// Kernel 1: q/k/v = W*x + b for one (b,h) row of 128 pixels, via WMMA.
// q,k written token-major [bn][n][c] (16B packed stores, feeds vectorized
// attention staging); v written channel-major [bn][c][n].
// ---------------------------------------------------------------------------
__global__ __launch_bounds__(256) void seblock_proj(
    const float* __restrict__ x,
    const float* __restrict__ wq, const float* __restrict__ bq,
    const float* __restrict__ wk, const float* __restrict__ bk,
    const float* __restrict__ wv, const float* __restrict__ bv,
    bf16* __restrict__ qt, bf16* __restrict__ kt, bf16* __restrict__ vt) {
  __shared__ __align__(16) bf16 Ws[3 * 64 * 72];   // [proj][c_out][c_in]
  __shared__ __align__(16) bf16 Xs[128 * 72];      // [pixel][c_in]  (X^T)
  __shared__ float Bs[192];

  const int t = threadIdx.x;
  const int b = blockIdx.x >> 6;   // /64
  const int h = blockIdx.x & 63;
  const float* Wp[3] = {wq, wk, wv};
  const float* bp[3] = {bq, bk, bv};

  for (int k = 0; k < 12; ++k) {                 // 3*64*64/4 = 3072 float4
    int idx = t + k * 256;
    int proj = idx >> 10;
    int rem = idx & 1023;
    int co = rem >> 4, ci4 = rem & 15;
    v4f wv4 = *(const v4f*)&Wp[proj][co * 64 + ci4 * 4];
    v4bf pk = {(bf16)wv4.x, (bf16)wv4.y, (bf16)wv4.z, (bf16)wv4.w};
    *(v4bf*)&Ws[(proj * 64 + co) * 72 + ci4 * 4] = pk;
  }
  if (t < 192) Bs[t] = bp[t >> 6][t & 63];
  for (int k = 0; k < 8; ++k) {                  // 64*128/4 = 2048 float4
    int idx = t + k * 256;
    int c = idx >> 5, wf4 = idx & 31;
    v4f xv = *(const v4f*)&x[(((size_t)b * 64 + c) * 64 + h) * 128 + wf4 * 4];
#pragma unroll
    for (int e = 0; e < 4; ++e)
      Xs[(wf4 * 4 + e) * 72 + c] = (bf16)xv[e];  // transpose scatter in LDS
  }
  __syncthreads();

  const int lane = t & 31;
  const int wave = t >> 5;
  const int lm = lane & 15;
  const int hi8 = (lane < 16) ? 0 : 8;
  const int pix = wave * 16 + lm;                // this lane's pixel column
  // forward block mapping: (b,h,pix) -> (bn, n)
  const int s = pix >> 6;
  const int wcol = pix & 63;
  const int jj = wcol >> 5, ww = wcol & 31;
  const int ii = h >> 5, hh = h & 31;
  const int bn = jj * 8 + ii * 4 + b;
  const int n = hh * 64 + ww * 2 + s;

  bf16* tokdst[2] = {qt, kt};
  v16bf bx0 = load_matB(Xs, pix, 0, 72, lane);
  v16bf bx1 = load_matB(Xs, pix, 32, 72, lane);
  for (int proj = 0; proj < 3; ++proj) {
    const bf16* Wbase = Ws + proj * 64 * 72;
    for (int ct = 0; ct < 4; ++ct) {
      v16bf a0 = load_matA(Wbase, ct * 16 + lm, 0, 72, lane);
      v16bf a1 = load_matA(Wbase, ct * 16 + lm, 32, 72, lane);
      v8f acc = {0.f, 0.f, 0.f, 0.f, 0.f, 0.f, 0.f, 0.f};
      acc = wmma_bf16(a0, bx0, acc);
      acc = wmma_bf16(a1, bx1, acc);
      if (proj < 2) {
        // token-major: rows r are consecutive channels -> one packed store
        v8bf w;
#pragma unroll
        for (int r = 0; r < 8; ++r)
          w[r] = (bf16)(acc[r] + Bs[proj * 64 + ct * 16 + r + hi8]);
        *(v8bf*)(tokdst[proj] + ((size_t)bn * 2048 + n) * 64 + ct * 16 + hi8) =
            w;
      } else {
        bf16* dst = vt + (size_t)bn * 64 * 2048 + n;
#pragma unroll
        for (int r = 0; r < 8; ++r) {
          int co = ct * 16 + r + hi8;
          dst[(size_t)co * 2048] = (bf16)(acc[r] + Bs[proj * 64 + co]);
        }
      }
    }
  }
}

// ---------------------------------------------------------------------------
// Kernel 2: fused flash attention per (block bn, 128-query tile).
// Each wave owns 16 queries; loops over 64-key tiles; online softmax.
// All global->LDS staging is 16B vectorized (b128 both sides).
// ---------------------------------------------------------------------------
__global__ __launch_bounds__(256) void seblock_attn(
    const bf16* __restrict__ qt, const bf16* __restrict__ kt,
    const bf16* __restrict__ vt, const float* __restrict__ x,
    float* __restrict__ out) {
  __shared__ __align__(16) bf16 Qs[128 * 72];     // [query][c]      (Q^T)
  __shared__ __align__(16) bf16 Ks[64 * 72];      // [key][c]        (K^T)
  __shared__ __align__(16) bf16 Vs[64 * 72];      // [c][key]        (V)
  __shared__ __align__(16) bf16 Ps[8 * 16 * 72];  // per-wave [n][m] (P)
  __shared__ float scl[8 * 16];
  __shared__ float rsum[8 * 16];

  const int t = threadIdx.x;
  const int bn = blockIdx.x >> 4;
  const int q0 = (blockIdx.x & 15) * 128;
  const bf16* qb = qt + (size_t)bn * 2048 * 64;   // [n][c]
  const bf16* kb = kt + (size_t)bn * 2048 * 64;   // [n][c]
  const bf16* vb = vt + (size_t)bn * 64 * 2048;   // [c][n]

  {  // Qs: 128 queries x 64c, fully vectorized (4x b128 per thread)
    int qi = t >> 1, c0 = (t & 1) * 32;
    const bf16* src = qb + ((size_t)(q0 + qi)) * 64 + c0;
#pragma unroll
    for (int e = 0; e < 4; ++e)
      *(v8bf*)&Qs[qi * 72 + c0 + e * 8] = *(const v8bf*)(src + e * 8);
  }

  const int lane = t & 31;
  const int wave = t >> 5;
  const int lm = lane & 15;
  const int hi8 = (lane < 16) ? 0 : 8;
  bf16* Pw = Ps + wave * 16 * 72;

  v8f acc[4];
  float rmax[8], rsm[8];
#pragma unroll
  for (int ct = 0; ct < 4; ++ct)
    acc[ct] = (v8f){0.f, 0.f, 0.f, 0.f, 0.f, 0.f, 0.f, 0.f};
#pragma unroll
  for (int r = 0; r < 8; ++r) {
    rmax[r] = -__builtin_inff();
    rsm[r] = 0.f;
  }

  __syncthreads();
  v16bf aq0 = load_matA(Qs, wave * 16 + lm, 0, 72, lane);   // fixed all loop
  v16bf aq1 = load_matA(Qs, wave * 16 + lm, 32, 72, lane);

  // per-thread staging coordinates (K and V tiles, 64x64 each)
  const int km = t >> 2, kc0 = (t & 3) * 16;     // K: row=key m, 16c chunk
  const int vc = t >> 2, vm0 = (t & 3) * 16;     // V: row=c, 16m chunk

  for (int m0 = 0; m0 < 2048; m0 += 64) {
    __syncthreads();
    {
      const bf16* ksrc = kb + ((size_t)(m0 + km)) * 64 + kc0;
      *(v8bf*)&Ks[km * 72 + kc0] = *(const v8bf*)(ksrc);
      *(v8bf*)&Ks[km * 72 + kc0 + 8] = *(const v8bf*)(ksrc + 8);
      const bf16* vsrc = vb + (size_t)vc * 2048 + m0 + vm0;
      *(v8bf*)&Vs[vc * 72 + vm0] = *(const v8bf*)(vsrc);
      *(v8bf*)&Vs[vc * 72 + vm0 + 8] = *(const v8bf*)(vsrc + 8);
    }
    __syncthreads();
    if (m0 + 64 < 2048) {  // stream next tile under the WMMA compute
      __builtin_prefetch(kb + ((size_t)(m0 + 64 + km)) * 64 + kc0, 0, 1);
      __builtin_prefetch(vb + (size_t)vc * 2048 + m0 + 64 + vm0, 0, 1);
    }

    // S[16n x 64m] = Q^T K  (rows n, cols m; reduce over c=64 in 2 chunks)
    v8f S[4];
#pragma unroll
    for (int mt = 0; mt < 4; ++mt) {
      v16bf b0 = load_matB(Ks, mt * 16 + lm, 0, 72, lane);
      v16bf b1 = load_matB(Ks, mt * 16 + lm, 32, 72, lane);
      v8f sacc = {0.f, 0.f, 0.f, 0.f, 0.f, 0.f, 0.f, 0.f};
      sacc = wmma_bf16(aq0, b0, sacc);
      sacc = wmma_bf16(aq1, b1, sacc);
      S[mt] = sacc;
    }

    // ---- online softmax (flash): per-row stats via 16-lane butterflies ----
    float tm[8], sc[8], psum[8];
#pragma unroll
    for (int r = 0; r < 8; ++r)
      tm[r] = fmaxf(fmaxf(S[0][r], S[1][r]), fmaxf(S[2][r], S[3][r]));
    for (int mask = 1; mask < 16; mask <<= 1)
#pragma unroll
      for (int r = 0; r < 8; ++r) tm[r] = fmaxf(tm[r], xshf(tm[r], mask));
#pragma unroll
    for (int r = 0; r < 8; ++r) {
      float nm = fmaxf(rmax[r], tm[r]);
      sc[r] = __expf(rmax[r] - nm);   // exp(-inf)=0 on first tile
      rmax[r] = nm;
      psum[r] = 0.f;
    }
#pragma unroll
    for (int mt = 0; mt < 4; ++mt)
#pragma unroll
      for (int r = 0; r < 8; ++r) {
        float p = __expf(S[mt][r] - rmax[r]);
        psum[r] += p;
        Pw[(r + hi8) * 72 + mt * 16 + lm] = (bf16)p;
      }
    for (int mask = 1; mask < 16; mask <<= 1)
#pragma unroll
      for (int r = 0; r < 8; ++r) psum[r] += xshf(psum[r], mask);
#pragma unroll
    for (int r = 0; r < 8; ++r) rsm[r] = rsm[r] * sc[r] + psum[r];

    // broadcast per-row rescale to all lanes by column (ctx cols are queries)
    if (lm == 0) {
#pragma unroll
      for (int r = 0; r < 8; ++r) scl[wave * 16 + r + hi8] = sc[r];
    }
    asm volatile("s_wait_dscnt 0" ::: "memory");  // wave-private LDS handshake
    float cscale = scl[wave * 16 + lm];
#pragma unroll
    for (int ct = 0; ct < 4; ++ct)
#pragma unroll
      for (int r = 0; r < 8; ++r) acc[ct][r] *= cscale;

    // ctx[64c x 16n] += V[c,m] * P^T[m,n]  (reduce over m=64 in 2 chunks)
#pragma unroll
    for (int mc = 0; mc < 64; mc += 32) {
      v16bf pb = load_matB(Pw, lm, mc, 72, lane);
#pragma unroll
      for (int ct = 0; ct < 4; ++ct) {
        v16bf av = load_matA(Vs, ct * 16 + lm, mc, 72, lane);
        acc[ct] = wmma_bf16(av, pb, acc[ct]);
      }
    }
  }

  // normalize, add residual, scatter back through the inverse block mapping
  if (lm == 0) {
#pragma unroll
    for (int r = 0; r < 8; ++r) rsum[wave * 16 + r + hi8] = rsm[r];
  }
  asm volatile("s_wait_dscnt 0" ::: "memory");
  float inv = 1.0f / rsum[wave * 16 + lm];

  const int ng = q0 + wave * 16 + lm;            // this lane's query (column)
  const int b = bn & 3, ii = (bn >> 2) & 1, jj = bn >> 3;
  const int hh = ng >> 6, ww = (ng >> 1) & 31, s = ng & 1;
  const int h = ii * 32 + hh, Wf = s * 64 + jj * 32 + ww;
  const size_t base = (size_t)b * 524288 + (size_t)h * 128 + Wf;
#pragma unroll
  for (int ct = 0; ct < 4; ++ct)
#pragma unroll
    for (int r = 0; r < 8; ++r) {
      int c = ct * 16 + r + hi8;
      size_t o = base + (size_t)c * 8192;
      out[o] = acc[ct][r] * inv + x[o];
    }
}

extern "C" void kernel_launch(void* const* d_in, const int* in_sizes, int n_in,
                              void* d_out, int out_size, void* d_ws,
                              size_t ws_size, hipStream_t stream) {
  const float* x = (const float*)d_in[0];
  const float* wq = (const float*)d_in[1];
  const float* bq = (const float*)d_in[2];
  const float* wk = (const float*)d_in[3];
  const float* bk = (const float*)d_in[4];
  const float* wv = (const float*)d_in[5];
  const float* bv = (const float*)d_in[6];
  float* out = (float*)d_out;

  bf16* qt = (bf16*)d_ws;                        // [16][2048][64]
  bf16* kt = qt + (size_t)16 * 2048 * 64;        // [16][2048][64]
  bf16* vt = kt + (size_t)16 * 2048 * 64;        // [16][64][2048]

  seblock_proj<<<256, 256, 0, stream>>>(x, wq, bq, wk, bk, wv, bv, qt, kt, vt);
  seblock_attn<<<256, 256, 0, stream>>>(qt, kt, vt, x, out);
}
